// GNN_Base_25752623907485
// MI455X (gfx1250) — matrix operand order repaired and checked by reading, not compiled, not run on hardware
//
#include <hip/hip_runtime.h>
#include <hip/hip_bf16.h>

// ---------------------------------------------------------------------------
// GINE GNN on gfx1250:
//   * bf16 WMMA (f32 accumulate) for all Linear layers
//   * async global->LDS copies (ASYNCcnt) double-buffering the A tiles
//   * VALU + f32 atomics for edge message/scatter
//   * BatchNorm stats fused into the GEMM epilogue
// ---------------------------------------------------------------------------

typedef __attribute__((ext_vector_type(16))) __bf16          v16bf;
typedef __attribute__((ext_vector_type(16))) unsigned short  v16us;
typedef __attribute__((ext_vector_type(8)))  float           v8f;

typedef int v4i __attribute__((vector_size(16)));     // int __vector(4)
typedef __attribute__((address_space(1))) v4i GV4;    // global v4i
typedef __attribute__((address_space(3))) v4i LV4;    // LDS v4i

#define N_NODES  200000
#define N_EDGES  800000
#define N_GRAPHS 4000
#define BN_EPS   1e-5f
#define MT       2          // M-tiles resident per LDS buffer (B-frag reuse)

__device__ __forceinline__ unsigned short f2bf(float f) {
    unsigned int u = __builtin_bit_cast(unsigned int, f);
    u += 0x7fffu + ((u >> 16) & 1u);          // round-to-nearest-even
    return (unsigned short)(u >> 16);
}

// ---- CDNA5 async global->LDS copy (16B per lane), ASYNCcnt tracked --------
__device__ __forceinline__ void async_cp16(const unsigned short* g,
                                           unsigned short* l) {
#if __has_builtin(__builtin_amdgcn_global_load_async_to_lds_b128)
    __builtin_amdgcn_global_load_async_to_lds_b128((GV4*)g, (LV4*)l, 0, 0);
#else
    unsigned lofs = (unsigned)(__UINTPTR_TYPE__)(LV4*)l;
    asm volatile("global_load_async_to_lds_b128 %0, %1, off"
                 :: "v"(lofs), "v"(g) : "memory");
#endif
}

__device__ __forceinline__ void wait_async0() {
#if __has_builtin(__builtin_amdgcn_s_wait_asynccnt)
    __builtin_amdgcn_s_wait_asynccnt(0);
#else
    asm volatile("s_wait_asynccnt 0x0" ::: "memory");
#endif
}

// ---------------------------------------------------------------------------
__global__ void zero_f32(float* __restrict__ p, long n) {
    long i = (long)blockIdx.x * blockDim.x + threadIdx.x;
    if (i < n) p[i] = 0.0f;
}

// f32 weight [K,N] -> bf16 transposed [N,K]  (contiguous-K B fragments)
__global__ void w_to_bf16_t(const float* __restrict__ w,
                            unsigned short* __restrict__ o, int K, int N) {
    int i = blockIdx.x * blockDim.x + threadIdx.x;
    if (i >= K * N) return;
    int k = i / N, n = i % N;
    o[(long)n * K + k] = f2bf(w[i]);
}

// GEMM A-operand pack: bf16(a [+ a2]) -> halves GEMM streaming bytes
__global__ void pack_bf16_add(const float* __restrict__ a,
                              const float* __restrict__ a2,
                              unsigned short* __restrict__ o, long n) {
    long i = (long)blockIdx.x * blockDim.x + threadIdx.x;
    if (i >= n) return;
    float v = a[i];
    if (a2) v += a2[i];
    o[i] = f2bf(v);
}

// ---------------------------------------------------------------------------
// Edge kernel: one wave per edge (bandwidth-bound; 23.3 TB/s HBM path).
//   e = edge_attr@We + be ; msg = relu(x[src] + e) ; agg[dst] += msg
// ---------------------------------------------------------------------------
__global__ __launch_bounds__(256)
void edge_scatter(const float* __restrict__ x,
                  const float* __restrict__ ea,
                  const int*   __restrict__ src,
                  const int*   __restrict__ dst,
                  const float* __restrict__ We,   // [16, C]
                  const float* __restrict__ be,   // [C]
                  float*       __restrict__ agg,  // [nodes, C]
                  int E, int C) {
    int gwave = (int)((blockIdx.x * (long)blockDim.x + threadIdx.x) >> 5);
    int lane  = threadIdx.x & 31;
    if (gwave >= E) return;
    int s = src[gwave], d = dst[gwave];

    float e[16];
#pragma unroll
    for (int k = 0; k < 16; ++k) e[k] = ea[(long)gwave * 16 + k];

    for (int c = lane; c < C; c += 32) {
        float acc = be[c];
#pragma unroll
        for (int k = 0; k < 16; ++k) acc = fmaf(e[k], We[(long)k * C + c], acc);
        float m = fmaxf(x[(long)s * C + c] + acc, 0.0f);
        atomicAdd(&agg[(long)d * C + c], m);
    }
}

// ---------------------------------------------------------------------------
// WMMA linear: out = act(A[M,K] @ W[K,N] + bias), A pre-packed bf16.
//   block = 8 waves; grid-stride over groups of MT M-tiles.
//   A tiles DMA'd to LDS via global_load_async_to_lds_b128, double-buffered:
//     s_wait_asynccnt 0 -> barrier -> issue next group -> WMMA on current.
//   Each wave owns one 16-col N-tile; B fragments loaded once per K-step and
//   reused across the MT resident M-tiles.
//   Optional: f32 out, bf16 out, fused relu, BN sum/sumsq atomics.
// Requires M%16==0, K%32==0, N%16==0, (M/16)%MT==0.
// ---------------------------------------------------------------------------
__global__ __launch_bounds__(256)
void wmma_linear(const unsigned short* __restrict__ Abf,   // [M,K] bf16
                 const unsigned short* __restrict__ WbfT,  // [N,K] bf16
                 const float* __restrict__ bias,           // [N]
                 float*          __restrict__ outf,        // nullable [M,N]
                 unsigned short* __restrict__ outbf,       // nullable [M,N]
                 int M, int K, int N, int relu,
                 float* __restrict__ stats) {              // nullable [2N]
    extern __shared__ unsigned short lds[];                // 2 * MT*16*K
    const int tileUS = MT * 16 * K;                        // ushorts per buffer
    const int Mtiles = M / 16;
    const int groups = Mtiles / MT;
    const int nb     = gridDim.x;

    const int wave   = threadIdx.x >> 5;
    const int lane   = threadIdx.x & 31;
    const int ntile  = blockIdx.y * 8 + wave;
    const bool active = (ntile * 16) < N;
    const int half = lane >> 4;
    const int row  = lane & 15;
    const int col  = ntile * 16 + (lane & 15);

    // ---- async stage of one MT-tile group into LDS buffer ----
    auto issue = [&](int grp, int buf) {
        const long gbase  = (long)grp * tileUS;            // ushort offset in Abf
        const int  chunks = tileUS / 8;                    // 16B per chunk
        unsigned short* l = &lds[buf * tileUS];
        for (int c = threadIdx.x; c < chunks; c += 256)
            async_cp16(Abf + gbase + (long)c * 8, l + c * 8);
    };

    if (blockIdx.x < groups) issue(blockIdx.x, 0);
    int buf = 0;
    for (int grp = blockIdx.x; grp < groups; grp += nb) {
        wait_async0();            // own wave's DMA done
        __syncthreads();          // all waves' DMA done; prev compute finished
        if (grp + nb < groups) issue(grp + nb, buf ^ 1);

        if (active) {
            v8f acc[MT];
#pragma unroll
            for (int t = 0; t < MT; ++t)
                acc[t] = (v8f){0.f, 0.f, 0.f, 0.f, 0.f, 0.f, 0.f, 0.f};

            const unsigned short* lbuf = &lds[buf * tileUS];
            for (int kt = 0; kt < K; kt += 32) {
                // B fragment (32x16): lanes 0-15 hold K=0..15, 16-31 hold 16..31
                v16us bus;
#pragma unroll
                for (int v = 0; v < 8; ++v) {
                    int kb = kt + 16 * half + 2 * v;
                    unsigned int bp =
                        *(const unsigned int*)&WbfT[(long)col * K + kb];
                    bus[2 * v]     = (unsigned short)(bp & 0xffffu);
                    bus[2 * v + 1] = (unsigned short)(bp >> 16);
                }
                v16bf b = __builtin_bit_cast(v16bf, bus);
#pragma unroll
                for (int t = 0; t < MT; ++t) {
                    // A fragment (16x32): VGPR v holds K pair
                    //   K=16*half+2v
                    v16us aus;
#pragma unroll
                    for (int v = 0; v < 8; ++v) {
                        int ka = kt + 16 * half + 2 * v;
                        unsigned int ap = *(const unsigned int*)
                            &lbuf[(t * 16 + row) * K + ka];
                        aus[2 * v]     = (unsigned short)(ap & 0xffffu);
                        aus[2 * v + 1] = (unsigned short)(ap >> 16);
                    }
                    v16bf a = __builtin_bit_cast(v16bf, aus);
                    acc[t] = __builtin_amdgcn_wmma_f32_16x16x32_bf16(
                        false, a, false, b, (short)0, acc[t], false, false);
                }
            }

            // ---- epilogue: bias, relu, stores, BN partial stats ----
            const float bv = bias ? bias[col] : 0.0f;
#pragma unroll
            for (int t = 0; t < MT; ++t) {
                int gt = grp * MT + t;
                if (gt >= Mtiles) continue;
                float s = 0.0f, sq = 0.0f;
#pragma unroll
                for (int v = 0; v < 8; ++v) {
                    int m = gt * 16 + half * 8 + v;
                    float val = acc[t][v] + bv;
                    if (relu) val = fmaxf(val, 0.0f);
                    if (outf)  outf[(long)m * N + col]  = val;
                    if (outbf) outbf[(long)m * N + col] = f2bf(val);
                    s += val;
                    sq += val * val;
                }
                if (stats) {
                    atomicAdd(&stats[col], s);
                    atomicAdd(&stats[N + col], sq);
                }
            }
        }
        buf ^= 1;
    }
}

// ---------------------------------------------------------------------------
// BatchNorm apply (training-mode biased stats)
// ---------------------------------------------------------------------------
__global__ void bn_apply(float* __restrict__ h, const float* __restrict__ stats,
                         const float* __restrict__ gamma,
                         const float* __restrict__ beta, int M, int N) {
    long i = (long)blockIdx.x * blockDim.x + threadIdx.x;
    if (i >= (long)M * N) return;
    int c = (int)(i % N);
    float inv = 1.0f / (float)M;
    float mu  = stats[c] * inv;
    float var = stats[N + c] * inv - mu * mu;
    h[i] = (h[i] - mu) * rsqrtf(var + BN_EPS) * gamma[c] + beta[c];
}

// ---------------------------------------------------------------------------
// global mean pool
// ---------------------------------------------------------------------------
__global__ void pool_scatter(const float* __restrict__ h,
                             const int* __restrict__ batch,
                             float* __restrict__ pool, float* __restrict__ cnt,
                             int M, int N) {
    long i = (long)blockIdx.x * blockDim.x + threadIdx.x;
    if (i >= (long)M * N) return;
    int n = (int)(i / N), c = (int)(i % N);
    int g = batch[n];
    atomicAdd(&pool[(long)g * N + c], h[i]);
    if (c == 0) atomicAdd(&cnt[g], 1.0f);
}

__global__ void pool_div_bf16(const float* __restrict__ pool,
                              const float* __restrict__ cnt,
                              unsigned short* __restrict__ obf, int G, int N) {
    long i = (long)blockIdx.x * blockDim.x + threadIdx.x;
    if (i >= (long)G * N) return;
    obf[i] = f2bf(pool[i] / fmaxf(cnt[i / N], 1.0f));
}

// ---------------------------------------------------------------------------
// final 32 -> 1 output linear
// ---------------------------------------------------------------------------
__global__ void head_out(const float* __restrict__ y, const float* __restrict__ Wo,
                         const float* __restrict__ bo, float* __restrict__ out,
                         int G) {
    int g = blockIdx.x * blockDim.x + threadIdx.x;
    if (g >= G) return;
    float acc = bo[0];
#pragma unroll
    for (int k = 0; k < 32; ++k) acc = fmaf(y[(long)g * 32 + k], Wo[k], acc);
    out[g] = acc;
}

// ---------------------------------------------------------------------------
// host orchestration
// ---------------------------------------------------------------------------
static inline long cdiv(long a, long b) { return (a + b - 1) / b; }

extern "C" void kernel_launch(void* const* d_in, const int* in_sizes, int n_in,
                              void* d_out, int out_size, void* d_ws, size_t ws_size,
                              hipStream_t stream) {
    (void)in_sizes; (void)n_in; (void)out_size; (void)ws_size;

    const float* x     = (const float*)d_in[0];
    const int*   eidx  = (const int*)d_in[1];
    const int*   batch = (const int*)d_in[2];
    const float* ea    = (const float*)d_in[3];
    const float *We[3], *be[3], *Wn[3], *bn[3], *g[3], *bt[3];
    for (int i = 0; i < 3; ++i) {
        We[i] = (const float*)d_in[4 + 6 * i];
        be[i] = (const float*)d_in[5 + 6 * i];
        Wn[i] = (const float*)d_in[6 + 6 * i];
        bn[i] = (const float*)d_in[7 + 6 * i];
        g[i]  = (const float*)d_in[8 + 6 * i];
        bt[i] = (const float*)d_in[9 + 6 * i];
    }
    const float* Wd0 = (const float*)d_in[22]; const float* bd0 = (const float*)d_in[23];
    const float* Wd1 = (const float*)d_in[24]; const float* bd1 = (const float*)d_in[25];
    const float* Wd2 = (const float*)d_in[26]; const float* bd2 = (const float*)d_in[27];
    const float* Wo  = (const float*)d_in[28]; const float* bo  = (const float*)d_in[29];
    float* outp = (float*)d_out;

    const int E = N_EDGES;
    const int* src = eidx;
    const int* dst = eidx + E;

    // ---- scratch carve-out ----
    char*  ws  = (char*)d_ws;
    size_t off = 0;
    auto carve = [&](size_t bytes) {
        void* p = ws + off;
        off += (bytes + 255) & ~(size_t)255;
        return p;
    };
    float* H0   = (float*)carve((size_t)N_NODES * 256 * 4);
    float* H1   = (float*)carve((size_t)N_NODES * 256 * 4);
    float* AGG  = (float*)carve((size_t)N_NODES * 256 * 4);
    unsigned short* ABF  = (unsigned short*)carve((size_t)N_NODES * 256 * 2);
    unsigned short* WBF  = (unsigned short*)carve((size_t)512 * 512 * 2);
    float* ST   = (float*)carve(1024 * 4);
    float* POOL = (float*)carve((size_t)N_GRAPHS * 256 * 4);
    float* CNT  = (float*)carve((size_t)N_GRAPHS * 4);
    unsigned short* PBF  = (unsigned short*)carve((size_t)N_GRAPHS * 256 * 2);
    unsigned short* Y1BF = (unsigned short*)carve((size_t)N_GRAPHS * 512 * 2);
    unsigned short* Y2BF = (unsigned short*)carve((size_t)N_GRAPHS * 128 * 2);
    float* Y3   = (float*)carve((size_t)N_GRAPHS * 32 * 4);

    const int TB = 256;
    auto zero = [&](float* p, long n) {
        zero_f32<<<dim3((unsigned)cdiv(n, TB)), TB, 0, stream>>>(p, n);
    };
    auto gemm = [&](const unsigned short* A, const float* W, const float* b,
                    float* of, unsigned short* ob, int M, int K, int N,
                    int relu, float* stats) {
        w_to_bf16_t<<<dim3((unsigned)cdiv((long)K * N, TB)), TB, 0, stream>>>(
            W, WBF, K, N);
        long groups = (M / 16) / MT;
        unsigned gx = (unsigned)(groups < 1024 ? groups : 1024);
        wmma_linear<<<dim3(gx, (unsigned)((N + 127) / 128)), TB,
                      (size_t)2 * MT * 16 * K * 2, stream>>>(
            A, WBF, b, of, ob, M, K, N, relu, stats);
    };

    // ---- 3 GINE conv layers ----
    const float* hin = x;
    float*       hbuf[2] = {H0, H1};
    for (int L = 0; L < 3; ++L) {
        const int cin  = (L == 0) ? 128 : 256;
        const int cout = 256;
        float* hout = hbuf[L & 1];

        zero(AGG, (long)N_NODES * cin);
        edge_scatter<<<dim3((unsigned)cdiv(E, 8)), TB, 0, stream>>>(
            hin, ea, src, dst, We[L], be[L], AGG, E, cin);

        // pack bf16(x + agg) once -> GEMM streams half the bytes
        pack_bf16_add<<<dim3((unsigned)cdiv((long)N_NODES * cin, TB)), TB, 0,
                        stream>>>(hin, AGG, ABF, (long)N_NODES * cin);

        zero(ST, 2 * cout);
        gemm(ABF, Wn[L], bn[L], hout, nullptr, N_NODES, cin, cout, 1, ST);

        bn_apply<<<dim3((unsigned)cdiv((long)N_NODES * cout, TB)), TB, 0, stream>>>(
            hout, ST, g[L], bt[L], N_NODES, cout);

        hin = hout;
    }

    // ---- global mean pool (bf16 output feeds dense head) ----
    zero(POOL, (long)N_GRAPHS * 256);
    zero(CNT, N_GRAPHS);
    pool_scatter<<<dim3((unsigned)cdiv((long)N_NODES * 256, TB)), TB, 0, stream>>>(
        hin, batch, POOL, CNT, N_NODES, 256);
    pool_div_bf16<<<dim3((unsigned)cdiv((long)N_GRAPHS * 256, TB)), TB, 0, stream>>>(
        POOL, CNT, PBF, N_GRAPHS, 256);

    // ---- dense head: bf16 activations between WMMA layers ----
    gemm(PBF,  Wd0, bd0, nullptr, Y1BF, N_GRAPHS, 256, 512, 1, nullptr);
    gemm(Y1BF, Wd1, bd1, nullptr, Y2BF, N_GRAPHS, 512, 128, 1, nullptr);
    gemm(Y2BF, Wd2, bd2, Y3, nullptr, N_GRAPHS, 128, 32, 1, nullptr);

    head_out<<<dim3((unsigned)cdiv(N_GRAPHS, TB)), TB, 0, stream>>>(
        Y3, Wo, bo, outp, N_GRAPHS);
}
